// DecoderLSTM_new_84241488543995
// MI455X (gfx1250) — matrix operand order repaired and checked by reading, not compiled, not run on hardware
//
#include <hip/hip_runtime.h>
#include <hip/hip_bf16.h>
#include <math.h>

typedef _Float16 f16;
typedef _Float16 v8h  __attribute__((ext_vector_type(8)));
typedef _Float16 v16h __attribute__((ext_vector_type(16)));
typedef float    v8f  __attribute__((ext_vector_type(8)));

#define BB 32
#define TT 20
#define PP 196
#define HH 512
#define VV 50000

// ---------------------------------------------------------------------------
// Generic WMMA GEMM: D[m][n] = sum_k A[m][k]*W[n][k] + bias[n]
// M = 32 rows (2 x 16 M-tiles), K multiple of 512, N multiple of 16.
// 256 threads = 8 waves, each wave owns one 16-wide N tile.
// grid.y tiles additional 32-row blocks of A/D (used for enc_proj).
// ---------------------------------------------------------------------------
__global__ void __launch_bounds__(256) gemm_wmma_f16(
    const f16* __restrict__ A, const f16* __restrict__ W,
    const float* __restrict__ bias, float* __restrict__ D,
    int N, int K, int ldd)
{
    __shared__ __align__(16) f16 As[32][520];  // 512 + 8 pad halves per row

    const int tid  = threadIdx.x;
    const int wave = tid >> 5;
    const int lane = tid & 31;
    const int hi   = lane >> 4;     // 0 for lanes 0-15, 1 for 16-31
    const int lm   = lane & 15;
    const int ntile = blockIdx.x * 8 + wave;
    const bool active = (ntile * 16) < N;
    const int col = ntile * 16 + lm;

    A += (size_t)blockIdx.y * 32 * K;
    D += (size_t)blockIdx.y * 32 * ldd;

    v8f c0 = {}; v8f c1 = {};
    const int nkb = K >> 9;  // K / 512
    for (int kb = 0; kb < nkb; ++kb) {
        // cooperative stage of A[0:32][kb*512 : kb*512+512] into LDS
        const f16* Ak = A + kb * 512;
        for (int i = tid; i < 32 * 64; i += 256) {
            int r = i >> 6, c8 = i & 63;
            *(v8h*)&As[r][c8 * 8] = *(const v8h*)&Ak[(size_t)r * K + c8 * 8];
        }
        __syncthreads();
        if (active) {
            const f16* Wk = W + (size_t)col * K + kb * 512;
#pragma unroll
            for (int kc = 0; kc < 16; ++kc) {
                // A fragment: lane holds row lm (and lm+16 for 2nd tile),
                // K halves {base..base+7, base+16..base+23}, base = hi*8
                int ka = kc * 32 + hi * 8;
                v8h a0lo = *(const v8h*)&As[lm][ka];
                v8h a0hi = *(const v8h*)&As[lm][ka + 16];
                v8h a1lo = *(const v8h*)&As[lm + 16][ka];
                v8h a1hi = *(const v8h*)&As[lm + 16][ka + 16];
                // B fragment: lane holds col, K halves {base..base+7, base+16..base+23}, base = hi*8
                const f16* wp = Wk + kc * 32 + hi * 8;
                v8h blo = *(const v8h*)wp;
                v8h bhi = *(const v8h*)(wp + 16);
                v16h a0, a1, b;
#pragma unroll
                for (int j = 0; j < 8; ++j) {
                    a0[j] = a0lo[j]; a0[j + 8] = a0hi[j];
                    a1[j] = a1lo[j]; a1[j + 8] = a1hi[j];
                    b[j]  = blo[j];  b[j + 8]  = bhi[j];
                }
                c0 = __builtin_amdgcn_wmma_f32_16x16x32_f16(false, a0, false, b,
                                                            (short)0, c0, false, false);
                c1 = __builtin_amdgcn_wmma_f32_16x16x32_f16(false, a1, false, b,
                                                            (short)0, c1, false, false);
            }
        }
        __syncthreads();
    }
    if (active) {
        float bv = bias[col];
#pragma unroll
        for (int r = 0; r < 8; ++r) {
            int m0 = r + hi * 8;                       // C/D layout: VGPR r -> M=r(+8)
            D[(size_t)m0 * ldd + col]        = c0[r] + bv;
            D[(size_t)(m0 + 16) * ldd + col] = c1[r] + bv;
        }
    }
}

// ---------------------------------------------------------------------------
// Attention: emb gather, scores = tanh(enc_proj + dp) . w_e, softmax, context.
// One block per batch element.
// ---------------------------------------------------------------------------
__global__ void __launch_bounds__(256) att_kernel(
    int t, const int* __restrict__ captions, const float* __restrict__ embedding,
    const float* __restrict__ enc, const float* __restrict__ enc_proj,
    const float* __restrict__ dp, const float* __restrict__ w_e,
    const float* __restrict__ b_e, f16* __restrict__ gatesA,
    float* __restrict__ attw_out)
{
    const int b = blockIdx.x, tid = threadIdx.x;
    __shared__ float s_dp[512];
    __shared__ float s_we[512];
    __shared__ float s_sc[224];
    __shared__ float red[256];

    int cap = captions[b * TT + t];
    for (int j = tid; j < 512; j += 256) {
        gatesA[b * 1536 + j] = (f16)embedding[(size_t)cap * HH + j];  // emb_t -> A[0:512)
        s_dp[j] = dp[b * HH + j];
        s_we[j] = w_e[j];
    }
    __syncthreads();

    const float* ep = enc_proj + (size_t)b * PP * HH;
    float sc = -1e30f;
    if (tid < PP) {
        float acc = 0.f;
        const float* row = ep + tid * HH;
        for (int d = 0; d < HH; ++d)
            acc += tanhf(row[d] + s_dp[d]) * s_we[d];
        sc = acc + b_e[0];
        s_sc[tid] = sc;
    }
    red[tid] = sc;
    __syncthreads();
    for (int s = 128; s > 0; s >>= 1) {
        if (tid < s) red[tid] = fmaxf(red[tid], red[tid + s]);
        __syncthreads();
    }
    float mx = red[0];
    __syncthreads();
    float ex = 0.f;
    if (tid < PP) ex = expf(s_sc[tid] - mx);
    red[tid] = ex;
    __syncthreads();
    for (int s = 128; s > 0; s >>= 1) {
        if (tid < s) red[tid] += red[tid + s];
        __syncthreads();
    }
    float inv = 1.f / red[0];
    __syncthreads();
    if (tid < PP) {
        float a = ex * inv;
        s_sc[tid] = a;
        attw_out[b * PP + tid] = a;
    }
    __syncthreads();

    const float* eb = enc + (size_t)b * PP * HH;
    for (int d = tid; d < HH; d += 256) {
        float acc = 0.f;
        for (int p = 0; p < PP; ++p)
            acc += s_sc[p] * eb[p * HH + d];
        gatesA[b * 1536 + 512 + d] = (f16)acc;                        // ctx -> A[512:1024)
    }
}

// ---------------------------------------------------------------------------
// LSTM pointwise: c' = sig(f)*c + sig(i)*tanh(g); h' = sig(o)*tanh(c')
// ---------------------------------------------------------------------------
__global__ void __launch_bounds__(256) point_kernel(
    const float* __restrict__ gates, float* __restrict__ c,
    f16* __restrict__ hf16, f16* __restrict__ gatesA)
{
    int idx = blockIdx.x * 256 + threadIdx.x;   // 0..16383
    int b = idx >> 9, j = idx & 511;
    const float* g = gates + (size_t)b * 2048;
    float gi = g[j], gf = g[j + 512], gg = g[j + 1024], go = g[j + 1536];
    float si = 1.f / (1.f + expf(-gi));
    float sf = 1.f / (1.f + expf(-gf));
    float so = 1.f / (1.f + expf(-go));
    float cn = sf * c[idx] + si * tanhf(gg);
    float hn = so * tanhf(cn);
    c[idx] = cn;
    f16 h16 = (f16)hn;
    hf16[idx] = h16;
    gatesA[b * 1536 + 1024 + j] = h16;                               // h -> A[1024:1536)
}

// ---------------------------------------------------------------------------
// init helpers
// ---------------------------------------------------------------------------
__global__ void cvt_flat(const float* __restrict__ s, f16* __restrict__ d, int n) {
    int i = blockIdx.x * 256 + threadIdx.x;
    if (i < n) d[i] = (f16)s[i];
}
__global__ void cvt_rows(const float* __restrict__ s, f16* __restrict__ d,
                         int srclen, int dstlen, int dstoff, int n) {
    int i = blockIdx.x * 256 + threadIdx.x;
    if (i < n) {
        int r = i / srclen, cc = i - r * srclen;
        d[(size_t)r * dstlen + dstoff + cc] = (f16)s[i];
    }
}
__global__ void bias_cat_kernel(const float* __restrict__ a, const float* __restrict__ b,
                                float* __restrict__ o, int n) {
    int i = blockIdx.x * 256 + threadIdx.x;
    if (i < n) o[i] = a[i] + b[i];
}
__global__ void zero_f32(float* p, int n) {
    int i = blockIdx.x * 256 + threadIdx.x;
    if (i < n) p[i] = 0.f;
}
__global__ void zero_f16(f16* p, int n) {
    int i = blockIdx.x * 256 + threadIdx.x;
    if (i < n) p[i] = (f16)0.f;
}
__global__ void zero_strided(float* p, int rows, int cols, size_t stride) {
    int i = blockIdx.x * 256 + threadIdx.x;
    int n = rows * cols;
    if (i < n) { int r = i / cols, cc = i - r * cols; p[(size_t)r * stride + cc] = 0.f; }
}

// ---------------------------------------------------------------------------
extern "C" void kernel_launch(void* const* d_in, const int* in_sizes, int n_in,
                              void* d_out, int out_size, void* d_ws, size_t ws_size,
                              hipStream_t stream)
{
    const float* enc       = (const float*)d_in[0];   // (32,196,512)
    const int*   captions  = (const int*)  d_in[1];   // (32,20)
    const float* embedding = (const float*)d_in[2];   // (50000,512)
    const float* W_ih      = (const float*)d_in[3];   // (2048,1024)
    const float* W_hh      = (const float*)d_in[4];   // (2048,512)
    const float* b_ih      = (const float*)d_in[5];
    const float* b_hh      = (const float*)d_in[6];
    const float* We        = (const float*)d_in[7];   // (512,512)
    const float* be        = (const float*)d_in[8];
    const float* Wd        = (const float*)d_in[9];   // (512,512)
    const float* bd        = (const float*)d_in[10];
    const float* w_e       = (const float*)d_in[11];  // (512,)
    const float* b_e       = (const float*)d_in[12];  // scalar
    const float* W_out     = (const float*)d_in[13];  // (50000,512)
    const float* b_out     = (const float*)d_in[14];

    float* out = (float*)d_out;                       // 32*20*50000 + 32*196

    // workspace layout
    char* w = (char*)d_ws;
    size_t off = 0;
    auto alloc = [&](size_t bytes) { char* p = w + off; off = (off + bytes + 255) & ~(size_t)255; return p; };
    f16*   Wout16   = (f16*)  alloc((size_t)VV * HH * 2);      // 51.2 MB
    f16*   Wcat16   = (f16*)  alloc((size_t)2048 * 1536 * 2);  // 6.3 MB
    f16*   Wd16     = (f16*)  alloc((size_t)512 * 512 * 2);
    f16*   We16     = (f16*)  alloc((size_t)512 * 512 * 2);
    f16*   enc16    = (f16*)  alloc((size_t)BB * PP * HH * 2); // 6.4 MB
    float* enc_proj = (float*)alloc((size_t)BB * PP * HH * 4); // 12.8 MB
    float* bias_cat = (float*)alloc(2048 * 4);
    float* dp       = (float*)alloc(BB * HH * 4);
    float* gates    = (float*)alloc(BB * 2048 * 4);
    f16*   gatesA   = (f16*)  alloc(BB * 1536 * 2);            // [emb|ctx|h] per row
    f16*   hf16     = (f16*)  alloc(BB * HH * 2);
    float* c_st     = (float*)alloc(BB * HH * 4);
    float* attw_ws  = (float*)alloc(BB * PP * 4);

    const int TPB = 256;
    auto blocks = [](int n) { return (n + 255) / 256; };

    // --- one-time setup ---
    cvt_flat<<<blocks(VV * HH), TPB, 0, stream>>>(W_out, Wout16, VV * HH);
    cvt_rows<<<blocks(2048 * 1024), TPB, 0, stream>>>(W_ih, Wcat16, 1024, 1536, 0,    2048 * 1024);
    cvt_rows<<<blocks(2048 * 512),  TPB, 0, stream>>>(W_hh, Wcat16, 512,  1536, 1024, 2048 * 512);
    cvt_flat<<<blocks(512 * 512), TPB, 0, stream>>>(Wd, Wd16, 512 * 512);
    cvt_flat<<<blocks(512 * 512), TPB, 0, stream>>>(We, We16, 512 * 512);
    cvt_flat<<<blocks(BB * PP * HH), TPB, 0, stream>>>(enc, enc16, BB * PP * HH);
    bias_cat_kernel<<<blocks(2048), TPB, 0, stream>>>(b_ih, b_hh, bias_cat, 2048);
    zero_f32<<<blocks(BB * HH), TPB, 0, stream>>>(c_st, BB * HH);
    zero_f16<<<blocks(BB * HH), TPB, 0, stream>>>(hf16, BB * HH);
    zero_f16<<<blocks(BB * 1536), TPB, 0, stream>>>(gatesA, BB * 1536);
    zero_strided<<<blocks(BB * VV), TPB, 0, stream>>>(out, BB, VV, (size_t)TT * VV); // outputs[:,0,:]=0

    // enc_proj = enc @ We.T + be  : 196 row-blocks of 32, N=512 -> grid (4,196)
    gemm_wmma_f16<<<dim3(4, PP), TPB, 0, stream>>>(enc16, We16, be, enc_proj, 512, 512, 512);

    // --- 19 sequential decode steps ---
    for (int t = 0; t < TT - 1; ++t) {
        // dp = h @ Wd.T + bd
        gemm_wmma_f16<<<dim3(4, 1), TPB, 0, stream>>>(hf16, Wd16, bd, dp, 512, 512, 512);
        // attention: scores/softmax/context + emb gather
        float* attw_dst = (t == TT - 2) ? (out + (size_t)BB * TT * VV) : attw_ws;
        att_kernel<<<dim3(BB), TPB, 0, stream>>>(t, captions, embedding, enc, enc_proj,
                                                 dp, w_e, b_e, gatesA, attw_dst);
        // gates = [emb,ctx] @ W_ih.T + h @ W_hh.T + (b_ih+b_hh)  (K=1536, N=2048)
        gemm_wmma_f16<<<dim3(16, 1), TPB, 0, stream>>>(gatesA, Wcat16, bias_cat, gates,
                                                       2048, 1536, 2048);
        // LSTM pointwise -> h,c
        point_kernel<<<dim3(64), TPB, 0, stream>>>(gates, c_st, hf16, gatesA);
        // out[:, t+1, :] = h @ W_out.T + b_out  (N=50000 -> 3125 tiles -> 391 blocks)
        gemm_wmma_f16<<<dim3(391, 1), TPB, 0, stream>>>(hf16, Wout16, b_out,
                                                        out + (size_t)(t + 1) * VV,
                                                        VV, 512, (size_t)TT * VV);
    }
}
